// Attention_29781303230506
// MI455X (gfx1250) — matrix operand order repaired and checked by reading, not compiled, not run on hardware
//
#include <hip/hip_runtime.h>

// Talking-heads attention for MI455X (gfx1250), bf16 WMMA pipeline.
// DIM=512, HEADS=8, DHEAD=64, SEQ=2048, BATCH=4.
//
// v2 changes vs v1:
//  - attention blocks now cover 32 query rows with 2 waves per head
//    (512 threads): halves K/V L2 traffic (2 GB -> 1 GB) at unchanged
//    per-wave register pressure.
//  - LDS logit/prob tiles padded to a 33-float row stride: the stage-3
//    post-mix read pattern (lane-stride 32 floats) was an 8-way bank
//    conflict; 33 is odd so all 16 lanes hit distinct banks.

#define DIMX   512
#define HEADS  8
#define DHEAD  64
#define INNERX 512
#define SEQ    2048
#define BATCH  4

#define JSTRIDE 33                 // padded row stride (floats) for 32-wide j tiles
#define HBLK    (16 * JSTRIDE)     // one head's 16x32 tile (528 floats)
#define SBLK    (HEADS * HBLK)     // one i-subtile's 8 heads (4224 floats)

typedef __attribute__((ext_vector_type(16))) __bf16 v16bf;
typedef __attribute__((ext_vector_type(8)))  __bf16 v8bf;
typedef __attribute__((ext_vector_type(8)))  float  v8f;

union BF16x16 { v16bf v; v8bf h[2]; };

static __device__ __forceinline__ __bf16 f2bf(float f) {
  unsigned u = __float_as_uint(f);
  u += 0x7FFFu + ((u >> 16) & 1u);           // round-to-nearest-even
  unsigned short s = (unsigned short)(u >> 16);
  __bf16 r;
  __builtin_memcpy(&r, &s, sizeof(r));
  return r;
}

static __device__ __forceinline__ v8f wmma_bf16(v16bf a, v16bf b, v8f c) {
  // v_wmma_f32_16x16x32_bf16  D = A(16x32) * B(32x16) + C
  return __builtin_amdgcn_wmma_f32_16x16x32_bf16(false, a, false, b,
                                                 (short)0, c, false, false);
}

// A-fragment (16-bit A 16x32): lane<16 holds row M=lane, K={lo..lo+7, hi..hi+7}
static __device__ __forceinline__ v16bf load_afrag(const __bf16* row, int lo, int hi) {
  BF16x16 u;
  u.h[0] = *reinterpret_cast<const v8bf*>(row + lo);
  u.h[1] = *reinterpret_cast<const v8bf*>(row + hi);
  return u.v;
}

static __device__ __forceinline__ float red_max16(float v) {
  v = fmaxf(v, __shfl_xor(v, 1, 16));
  v = fmaxf(v, __shfl_xor(v, 2, 16));
  v = fmaxf(v, __shfl_xor(v, 4, 16));
  v = fmaxf(v, __shfl_xor(v, 8, 16));
  return v;
}
static __device__ __forceinline__ float red_sum16(float v) {
  v += __shfl_xor(v, 1, 16);
  v += __shfl_xor(v, 2, 16);
  v += __shfl_xor(v, 4, 16);
  v += __shfl_xor(v, 8, 16);
  return v;
}
static __device__ __forceinline__ float red_sum32(float v) {
  v += __shfl_xor(v, 1, 32);
  v += __shfl_xor(v, 2, 32);
  v += __shfl_xor(v, 4, 32);
  v += __shfl_xor(v, 8, 32);
  v += __shfl_xor(v, 16, 32);
  return v;
}

// ---------------------------------------------------------------------------
// 1) LayerNorm (fp32 in -> bf16 out). One wave per row of 512.
// ---------------------------------------------------------------------------
__global__ __launch_bounds__(256) void ln_kernel(const float* __restrict__ x,
                                                 const float* __restrict__ gamma,
                                                 const float* __restrict__ beta,
                                                 __bf16* __restrict__ xn) {
  const int lane = threadIdx.x & 31, wave = threadIdx.x >> 5;
  const int row = blockIdx.x * 8 + wave;
  const float* xr = x + (size_t)row * DIMX;
  float v[16];
  float s = 0.f, sq = 0.f;
#pragma unroll
  for (int t = 0; t < 4; ++t) {
    float4 f = *reinterpret_cast<const float4*>(xr + lane * 16 + t * 4);
    v[t * 4 + 0] = f.x; v[t * 4 + 1] = f.y; v[t * 4 + 2] = f.z; v[t * 4 + 3] = f.w;
    s  += f.x + f.y + f.z + f.w;
    sq += f.x * f.x + f.y * f.y + f.z * f.z + f.w * f.w;
  }
  s  = red_sum32(s);
  sq = red_sum32(sq);
  const float mean = s * (1.f / DIMX);
  const float var  = sq * (1.f / DIMX) - mean * mean;
  const float rstd = rsqrtf(var + 1e-5f);
  __bf16* orow = xn + (size_t)row * DIMX;
#pragma unroll
  for (int e = 0; e < 16; ++e) {
    const int col = lane * 16 + e;
    orow[col] = f2bf((v[e] - mean) * rstd * gamma[col] + beta[col]);
  }
}

// ---------------------------------------------------------------------------
// 2) Weight transpose fp32 W[R][C] -> bf16 WT[C][R]
// ---------------------------------------------------------------------------
__global__ __launch_bounds__(256) void wt_kernel(const float* __restrict__ W,
                                                 __bf16* __restrict__ WT,
                                                 int R, int C) {
  const int idx = blockIdx.x * 256 + threadIdx.x;
  const int r = idx / C, c = idx % C;
  WT[(size_t)c * R + r] = f2bf(W[(size_t)r * C + c]);
}

// ---------------------------------------------------------------------------
// 3) QKV projection GEMM: xn(8192x512) @ [Wq|Wkv](512x1536), WMMA bf16.
//    q,k stored [b][h][n][64]; v stored transposed [b][h][64][n].
//    Each wave: one 16-row x 64-col strip; 8 waves/block.
// ---------------------------------------------------------------------------
__global__ __launch_bounds__(256) void qkv_kernel(const __bf16* __restrict__ xn,
                                                  const __bf16* __restrict__ WqT,
                                                  const __bf16* __restrict__ WkvT,
                                                  __bf16* __restrict__ q,
                                                  __bf16* __restrict__ k,
                                                  __bf16* __restrict__ vT) {
  const int lane = threadIdx.x & 31, wave = threadIdx.x >> 5;
  const int n = lane & 15, hs = lane >> 4;
  const int rt = blockIdx.y * 8 + wave;  // row tile 0..511
  const int r0 = rt * 16;
  const int c0 = blockIdx.x * 64;        // col group 0..23 -> cols c0..c0+63

  const v8f z8 = {0.f, 0.f, 0.f, 0.f, 0.f, 0.f, 0.f, 0.f};
  v8f acc[4] = {z8, z8, z8, z8};

  const __bf16* Brow[4];
#pragma unroll
  for (int t = 0; t < 4; ++t) {
    const int col = c0 + t * 16 + n;
    Brow[t] = (col < 512) ? (WqT + (size_t)col * DIMX)
                          : (WkvT + (size_t)(col - 512) * DIMX);
  }
  const __bf16* Arow = xn + (size_t)(r0 + n) * DIMX;

  for (int kk = 0; kk < DIMX; kk += 32) {
    const v16bf a = load_afrag(Arow, kk + hs * 16, kk + hs * 16 + 8);
#pragma unroll
    for (int t = 0; t < 4; ++t) {
      const v16bf b = *reinterpret_cast<const v16bf*>(Brow[t] + kk + hs * 16);
      acc[t] = wmma_bf16(a, b, acc[t]);
    }
  }

#pragma unroll
  for (int t = 0; t < 4; ++t) {
    const int col = c0 + t * 16 + n;  // 0..1535
#pragma unroll
    for (int r = 0; r < 8; ++r) {
      const int rr = r0 + hs * 8 + r;
      const int bI = rr >> 11, tok = rr & (SEQ - 1);
      const float val = acc[t][r];
      if (col < 512) {
        const int h = col >> 6, d = col & 63;
        q[(((size_t)(bI * HEADS + h)) * SEQ + tok) * DHEAD + d] = f2bf(val);
      } else if (col < 1024) {
        const int c = col - 512, h = c >> 6, d = c & 63;
        k[(((size_t)(bI * HEADS + h)) * SEQ + tok) * DHEAD + d] = f2bf(val);
      } else {
        const int c = col - 1024, h = c >> 6, d = c & 63;
        vT[(((size_t)(bI * HEADS + h)) * DHEAD + d) * SEQ + tok] = f2bf(val);
      }
    }
  }
}

// ---------------------------------------------------------------------------
// 4) Fused talking-heads attention. Block = 16 waves (512 threads):
//    wave w -> head g = w&7, i-subtile s = w>>3; block covers 32 query rows
//    of one batch. Two-pass online softmax: pass A computes per-row max/sum
//    of the pre-mixed logits, pass B recomputes logits, applies post-mix and
//    accumulates P@V via WMMA. The two s-waves of a head issue identical K/V
//    fragment addresses -> WGP$ coalesces them (halves L2 traffic).
// ---------------------------------------------------------------------------
__global__ __launch_bounds__(512) void attn_kernel(const __bf16* __restrict__ q,
                                                   const __bf16* __restrict__ k,
                                                   const __bf16* __restrict__ vT,
                                                   const float* __restrict__ mix_pre,
                                                   const float* __restrict__ mix_post,
                                                   __bf16* __restrict__ attnout) {
  __shared__ float Dls[2 * SBLK];  // raw dots, 2 subtiles x 8 heads  (33.0 KB)
  __shared__ float Pls[2 * SBLK];  // exp'd probs                      (33.0 KB)

  const int lane = threadIdx.x & 31, wave = threadIdx.x >> 5;
  const int n = lane & 15, hs = lane >> 4;
  const int b = blockIdx.y;
  const int g = wave & 7;          // this wave's head
  const int s = wave >> 3;         // i-subtile 0/1
  const int i0 = blockIdx.x * 32 + s * 16;

  const __bf16* qh = q  + (((size_t)(b * HEADS + g)) * SEQ + i0) * DHEAD;
  const __bf16* kh = k  +  ((size_t)(b * HEADS + g)) * SEQ * DHEAD;
  const __bf16* vh = vT +  ((size_t)(b * HEADS + g)) * DHEAD * SEQ;

  float* Dsub = Dls + s * SBLK;    // this subtile's 8-head logit block
  float* Psub = Pls + s * SBLK;

  // q A-fragments for the two K-chunks d0 = 0, 32 (kept in registers).
  v16bf aq[2];
#pragma unroll
  for (int di = 0; di < 2; ++di) {
    const int d0 = di * 32;
    aq[di] = load_afrag(qh + (size_t)n * DHEAD, d0 + hs * 16, d0 + hs * 16 + 8);
  }

  float mp[HEADS], mq[HEADS];
#pragma unroll
  for (int h = 0; h < HEADS; ++h) {
    mp[h] = mix_pre[h * HEADS + g];
    mq[h] = mix_post[h * HEADS + g];
  }

  float m[8], l[8];
#pragma unroll
  for (int r = 0; r < 8; ++r) { m[r] = -1e30f; l[r] = 0.f; }

  // stage 1: raw dots (this wave's head, its 16 rows) for a 16x32 j-tile
  auto compute_dots = [&](int j0) {
    if (j0 + 32 < SEQ) {
      __builtin_prefetch(kh + (size_t)(j0 + 32 + n) * DHEAD, 0, 1);  // global_prefetch
      __builtin_prefetch(vh + (size_t)n * SEQ + j0 + 32, 0, 1);
    }
    float* Dmy = Dsub + g * HBLK;
#pragma unroll
    for (int jsI = 0; jsI < 2; ++jsI) {
      const int js = jsI * 16;
      v8f c = {0.f, 0.f, 0.f, 0.f, 0.f, 0.f, 0.f, 0.f};
#pragma unroll
      for (int di = 0; di < 2; ++di) {
        const int d0 = di * 32;
        // B(32x16): lane<16 -> K rows d0..d0+15 of column j = n (contiguous 32B)
        const v16bf bk = *reinterpret_cast<const v16bf*>(
            kh + (size_t)(j0 + js + n) * DHEAD + d0 + hs * 16);
        c = wmma_bf16(aq[di], bk, c);
      }
#pragma unroll
      for (int r = 0; r < 8; ++r)
        Dmy[(r + hs * 8) * JSTRIDE + js + n] = c[r] * 0.125f;  // *1/sqrt(64)
    }
  };

  // ---------------- PASS A: softmax statistics ----------------
  for (int j0 = 0; j0 < SEQ; j0 += 32) {
    compute_dots(j0);
    __syncthreads();
#pragma unroll
    for (int r = 0; r < 8; ++r) {
      const int i = r + hs * 8;
      float s0 = 0.f, s1 = 0.f;
#pragma unroll
      for (int h = 0; h < HEADS; ++h) {
        s0 += mp[h] * Dsub[h * HBLK + i * JSTRIDE + n];
        s1 += mp[h] * Dsub[h * HBLK + i * JSTRIDE + 16 + n];
      }
      float tm = red_max16(fmaxf(s0, s1));
      const float mn = fmaxf(m[r], tm);
      float se = red_sum16(__expf(s0 - mn) + __expf(s1 - mn));
      l[r] = l[r] * __expf(m[r] - mn) + se;
      m[r] = mn;
    }
    __syncthreads();
  }

  // ---------------- PASS B: P@V with post-mix ----------------
  const v8f z8 = {0.f, 0.f, 0.f, 0.f, 0.f, 0.f, 0.f, 0.f};
  v8f O[4] = {z8, z8, z8, z8};

  for (int j0 = 0; j0 < SEQ; j0 += 32) {
    compute_dots(j0);
    __syncthreads();
    // stage 2: pre-mix + exp (unnormalized), this wave's head g -> Psub[g]
#pragma unroll
    for (int r = 0; r < 8; ++r) {
      const int i = r + hs * 8;
      float s0 = 0.f, s1 = 0.f;
#pragma unroll
      for (int h = 0; h < HEADS; ++h) {
        s0 += mp[h] * Dsub[h * HBLK + i * JSTRIDE + n];
        s1 += mp[h] * Dsub[h * HBLK + i * JSTRIDE + 16 + n];
      }
      Psub[g * HBLK + i * JSTRIDE + n]      = __expf(s0 - m[r]);
      Psub[g * HBLK + i * JSTRIDE + 16 + n] = __expf(s1 - m[r]);
    }
    __syncthreads();
    // stage 3: post-mix into an A-fragment (16x32, A-layout), then P@V.
    // Lane-stride JSTRIDE=33 floats -> all 16 lanes hit distinct LDS banks.
    v16bf av;
#pragma unroll
    for (int t = 0; t < 8; ++t) {
      const int j = hs * 16 + t;  // lane<16: K 0..15; lane>=16: K 16..31
      float a0 = 0.f, a1 = 0.f;
#pragma unroll
      for (int h = 0; h < HEADS; ++h) {
        a0 += mq[h] * Psub[h * HBLK + n * JSTRIDE + j];
        a1 += mq[h] * Psub[h * HBLK + n * JSTRIDE + 8 + j];
      }
      av[t]     = f2bf(a0);
      av[8 + t] = f2bf(a1);
    }
#pragma unroll
    for (int t = 0; t < 4; ++t) {
      const int d0 = t * 16;
      // B(32x16): lane<16 -> K rows j0..j0+15 of column d = d0+n (contiguous, vT layout)
      const v16bf bv = *reinterpret_cast<const v16bf*>(
          vh + (size_t)(d0 + n) * SEQ + j0 + hs * 16);
      O[t] = wmma_bf16(av, bv, O[t]);
    }
    __syncthreads();
  }

  // epilogue: normalize by l and store [b][n][h*64+d] bf16
#pragma unroll
  for (int t = 0; t < 4; ++t) {
#pragma unroll
    for (int r = 0; r < 8; ++r) {
      const int i = r + hs * 8;
      const int d = t * 16 + n;
      const float val = O[t][r] / l[r];
      attnout[(size_t)(b * SEQ + i0 + i) * INNERX + g * DHEAD + d] = f2bf(val);
    }
  }
}

// ---------------------------------------------------------------------------
// 5) Output projection: attnout(8192x512)bf16 @ Wout(512x512) + bout -> f32
// ---------------------------------------------------------------------------
__global__ __launch_bounds__(256) void outproj_kernel(const __bf16* __restrict__ ao,
                                                      const __bf16* __restrict__ WoutT,
                                                      const float* __restrict__ bout,
                                                      float* __restrict__ out) {
  const int lane = threadIdx.x & 31, wave = threadIdx.x >> 5;
  const int n = lane & 15, hs = lane >> 4;
  const int rt = blockIdx.y * 8 + wave;
  const int r0 = rt * 16;
  const int c0 = blockIdx.x * 64;

  const v8f z8 = {0.f, 0.f, 0.f, 0.f, 0.f, 0.f, 0.f, 0.f};
  v8f acc[4] = {z8, z8, z8, z8};

  const __bf16* Arow = ao + (size_t)(r0 + n) * INNERX;
  for (int kk = 0; kk < INNERX; kk += 32) {
    const v16bf a = load_afrag(Arow, kk + hs * 16, kk + hs * 16 + 8);
#pragma unroll
    for (int t = 0; t < 4; ++t) {
      const int col = c0 + t * 16 + n;
      const v16bf b = *reinterpret_cast<const v16bf*>(WoutT + (size_t)col * INNERX + kk + hs * 16);
      acc[t] = wmma_bf16(a, b, acc[t]);
    }
  }
#pragma unroll
  for (int t = 0; t < 4; ++t) {
    const int col = c0 + t * 16 + n;
    const float bb = bout[col];
#pragma unroll
    for (int r = 0; r < 8; ++r) {
      const int rr = r0 + hs * 8 + r;
      out[(size_t)rr * DIMX + col] = acc[t][r] + bb;
    }
  }
}

// ---------------------------------------------------------------------------
extern "C" void kernel_launch(void* const* d_in, const int* in_sizes, int n_in,
                              void* d_out, int out_size, void* d_ws, size_t ws_size,
                              hipStream_t stream) {
  const float* x        = (const float*)d_in[0];
  const float* gamma    = (const float*)d_in[1];
  const float* beta     = (const float*)d_in[2];
  const float* Wq       = (const float*)d_in[3];
  const float* Wkv      = (const float*)d_in[4];
  const float* mix_pre  = (const float*)d_in[5];
  const float* mix_post = (const float*)d_in[6];
  const float* Wout     = (const float*)d_in[7];
  const float* bout     = (const float*)d_in[8];
  float* out = (float*)d_out;

  // Workspace layout (bytes), ~42 MiB total, all 256B-aligned.
  char* ws = (char*)d_ws;
  __bf16* xn    = (__bf16*)(ws + 0);          // 8192*512*2      = 8 MiB
  __bf16* qb    = (__bf16*)(ws + 8388608);    // 4*8*2048*64*2   = 8 MiB
  __bf16* kb    = (__bf16*)(ws + 16777216);   // 8 MiB
  __bf16* vTb   = (__bf16*)(ws + 25165824);   // 8 MiB  ([b][h][64][n])
  __bf16* ao    = (__bf16*)(ws + 33554432);   // 8 MiB
  __bf16* WqT   = (__bf16*)(ws + 41943040);   // 512*512*2  = 0.5 MiB
  __bf16* WkvT  = (__bf16*)(ws + 42467328);   // 1024*512*2 = 1 MiB
  __bf16* WoutT = (__bf16*)(ws + 43515904);   // 0.5 MiB

  ln_kernel<<<1024, 256, 0, stream>>>(x, gamma, beta, xn);
  wt_kernel<<<1024, 256, 0, stream>>>(Wq, WqT, 512, 512);
  wt_kernel<<<2048, 256, 0, stream>>>(Wkv, WkvT, 512, 1024);
  wt_kernel<<<1024, 256, 0, stream>>>(Wout, WoutT, 512, 512);
  qkv_kernel<<<dim3(24, 64), 256, 0, stream>>>(xn, WqT, WkvT, qb, kb, vTb);
  attn_kernel<<<dim3(SEQ / 32, BATCH), 512, 0, stream>>>(qb, kb, vTb, mix_pre, mix_post, ao);
  outproj_kernel<<<dim3(8, 64), 256, 0, stream>>>(ao, WoutT, bout, out);
  (void)in_sizes; (void)n_in; (void)out_size; (void)ws_size;
}
